// SwinBlock3D_55044300866252
// MI455X (gfx1250) — matrix operand order repaired and checked by reading, not compiled, not run on hardware
//
#include <hip/hip_runtime.h>

// ---------------------------------------------------------------------------
// Types / helpers for CDNA5 (gfx1250) wave32 WMMA bf16 path
// ---------------------------------------------------------------------------
typedef __bf16 bf16_t;
typedef __bf16 v8bf  __attribute__((ext_vector_type(8)));
typedef __bf16 v16bf __attribute__((ext_vector_type(16)));
typedef float  v8f   __attribute__((ext_vector_type(8)));

#define DEV __device__ __forceinline__

// native f32 -> bf16 convert (gfx1250 has hardware bf16 cvt; RNE)
DEV bf16_t f2bf(float f) { return (bf16_t)f; }

// A fragment (16x32 bf16, row-major source, row = lane&15):
// lane half hi: j<8 -> K = koff..koff+7 ; j>=8 -> K = koff+16..koff+23
// caller passes koff = kbase + 8*hi
DEV v16bf frag_a(const bf16_t* rowp, int koff) {
  v8bf lo = *(const v8bf*)(rowp + koff);
  v8bf hi = *(const v8bf*)(rowp + koff + 16);
  return __builtin_shufflevector(lo, hi, 0,1,2,3,4,5,6,7,8,9,10,11,12,13,14,15);
}
// B fragment (32x16 bf16) loaded from TRANSPOSED weights (row = out-col = lane&15):
// same K order as A: j<8 -> koff..koff+7 ; j>=8 -> koff+16..koff+23; caller passes koff = kbase + 8*hi
DEV v16bf frag_b(const bf16_t* rowp, int koff) {
  v8bf lo = *(const v8bf*)(rowp + koff);
  v8bf hi = *(const v8bf*)(rowp + koff + 16);
  return __builtin_shufflevector(lo, hi, 0,1,2,3,4,5,6,7,8,9,10,11,12,13,14,15);
}

DEV v8f wmma_bf(v16bf a, v16bf b, v8f c) {
  return __builtin_amdgcn_wmma_f32_16x16x32_bf16(false, a, false, b, (short)0, c, false, false);
}

// ---------------------------------------------------------------------------
// Problem constants (B=1, T=16, H=W=112, C=96, heads=3, hd=32, win=2x7x7)
// ---------------------------------------------------------------------------
// LDS strides (elements). Chosen: byte stride multiple of 16 (aligned b128),
// and (bytes/4) mod 64 has gcd 4 with 64 -> conflict-free row striping.
constexpr int XS_  = 104;  // window tokens (bf16)     112 x 104
constexpr int QS_  = 104;  // Q (bf16)                 112 x 104
constexpr int KS_  = 104;  // K (bf16)                 112 x 104
constexpr int VTS_ = 136;  // V^T (bf16)                96 x 136 (k padded ->128, zeroed)
constexpr int SST_ = 100;  // logits (f32)              98 x 100
constexpr int PS_  = 136;  // probabilities (bf16)     112 x 136 (cols 98..127 zero)
constexpr int WQS_ = 104;  // qkv weights^T half (bf16)144 x 104
constexpr int WPS_ = 104;  // proj weights^T (bf16)     96 x 104
constexpr int OHS_ = 40;   // per-head attn out (bf16) 112 x 40

constexpr int OFF_Q   = 0;
constexpr int OFF_K   = OFF_Q  + 112 * QS_ * 2;   // 23296
constexpr int OFF_VT  = OFF_K  + 112 * KS_ * 2;   // 46592
constexpr int OFF_U1  = OFF_VT + 96 * VTS_ * 2;   // 72704 : Wqkv-half / logits / OH
constexpr int OFF_REG = OFF_U1 + 98 * SST_ * 4;   // 111904
constexpr int OFF_U2  = OFF_REG + 112 * 4;        // 112352 : X / P
constexpr int OFF_WP  = OFF_U2 + 112 * PS_ * 2;   // 142816
constexpr int ATTN_LDS = OFF_WP + 96 * WPS_ * 2;  // 162784 (<160KB)
static_assert(ATTN_LDS <= 163840, "attn LDS");
static_assert((OFF_U2 & 15) == 0 && (OFF_WP & 15) == 0 && (OFF_U1 & 15) == 0, "align");

// ---------------------------------------------------------------------------
// Kernel 1: window attention. 1 block = 1 window (2048 blocks, 256 threads).
// Writes x1 = x + attn_proj into `out`.
// ---------------------------------------------------------------------------
__global__ __launch_bounds__(256, 1)
void swin_attn(const float* __restrict__ x,  const float* __restrict__ g1,
               const float* __restrict__ b1, const float* __restrict__ qkv_w,
               const float* __restrict__ qkv_b, const float* __restrict__ rpb,
               const float* __restrict__ proj_w, const float* __restrict__ proj_b,
               float* __restrict__ out)
{
  extern __shared__ __align__(16) char smem[];
  bf16_t* Qs = (bf16_t*)(smem + OFF_Q);
  bf16_t* Ks = (bf16_t*)(smem + OFF_K);
  bf16_t* Vt = (bf16_t*)(smem + OFF_VT);
  bf16_t* Wq = (bf16_t*)(smem + OFF_U1);   // phase 1: qkv weight half (transposed)
  float*  Sb = (float*) (smem + OFF_U1);   // per-head logits (aliases Wq, later)
  bf16_t* OH = (bf16_t*)(smem + OFF_U1);   // per-head attn-out (aliases Sb, after softmax)
  int*    regS = (int*)  (smem + OFF_REG);
  bf16_t* Xs = (bf16_t*)(smem + OFF_U2);   // LN'd window tokens
  bf16_t* Pb = (bf16_t*)(smem + OFF_U2);   // probabilities (aliases Xs after QKV)
  bf16_t* Wp = (bf16_t*)(smem + OFF_WP);   // proj weights (transposed)

  const int tid  = threadIdx.x;
  const int wave = tid >> 5, lane = tid & 31, l = lane & 15, hi2 = lane >> 4;
  const int wi = blockIdx.x;
  const int tb = wi >> 8, hb = (wi >> 4) & 15, wb = wi & 15;

  // warm L2 for per-block weight re-reads (global_prefetch_b8)
  if (tid == 0) {
    __builtin_prefetch(qkv_w, 0, 3);
    __builtin_prefetch(proj_w, 0, 3);
    __builtin_prefetch(rpb, 0, 3);
  }

  // ---------------- Phase 0: LN1 + shifted-window gather -> Xs (bf16) -------
  for (int t = wave; t < 112; t += 8) {
    if (t < 98) {
      int it = t / 49, r2 = t % 49, ih = r2 / 7, iw = r2 % 7;
      int ts = (tb * 2 + it + 1) & 15;                      // roll(-1) on T=16
      int hs = hb * 7 + ih + 3; if (hs >= 112) hs -= 112;   // roll(-3) on H
      int vs = wb * 7 + iw + 3; if (vs >= 112) vs -= 112;   // roll(-3) on W
      int base = ((ts * 112 + hs) * 112 + vs) * 96;
      float v0 = x[base + lane], v1 = x[base + lane + 32], v2 = x[base + lane + 64];
      float s = v0 + v1 + v2, q = v0 * v0 + v1 * v1 + v2 * v2;
      for (int o = 16; o >= 1; o >>= 1) { s += __shfl_xor(s, o); q += __shfl_xor(q, o); }
      float mean = s * (1.f / 96.f);
      float rstd = rsqrtf(q * (1.f / 96.f) - mean * mean + 1e-5f);
      Xs[t * XS_ + lane]      = f2bf((v0 - mean) * rstd * g1[lane]      + b1[lane]);
      Xs[t * XS_ + lane + 32] = f2bf((v1 - mean) * rstd * g1[lane + 32] + b1[lane + 32]);
      Xs[t * XS_ + lane + 64] = f2bf((v2 - mean) * rstd * g1[lane + 64] + b1[lane + 64]);
    } else {  // zero-pad rows 98..111 so padded QKV rows stay finite
      Xs[t * XS_ + lane] = f2bf(0.f);
      Xs[t * XS_ + lane + 32] = f2bf(0.f);
      Xs[t * XS_ + lane + 64] = f2bf(0.f);
    }
  }
  // zero Vt K-pad columns 112..127 (multiplied by P==0; must not be NaN)
  for (int i = tid; i < 96 * 2; i += 256) {
    v8bf z = {};
    *(v8bf*)(Vt + (i >> 1) * VTS_ + 112 + 8 * (i & 1)) = z;
  }
  // qkv weights, half 0 (output cols 0..143), transposed: Wq[n][k] = qkv_w[k][n]
  for (int i = tid; i < 96 * 144; i += 256) {
    int k = i / 144, n = i % 144;
    Wq[n * WQS_ + k] = f2bf(qkv_w[k * 288 + n]);
  }
  // proj weights, transposed: Wp[n][k] = proj_w[k][n]
  for (int i = tid; i < 96 * 96; i += 256) {
    int k = i / 96, n = i % 96;
    Wp[n * WPS_ + k] = f2bf(proj_w[i]);
  }
  // shift-mask region id per token
  if (tid < 98) {
    int it = tid / 49, r2 = tid % 49, ih = r2 / 7, iw = r2 % 7;
    int at = tb * 2 + it, ah = hb * 7 + ih, aw = wb * 7 + iw;
    regS[tid] = ((at < 14) ? 0 : ((at < 15) ? 1 : 2)) * 9 +
                ((ah < 105) ? 0 : ((ah < 109) ? 1 : 2)) * 3 +
                ((aw < 105) ? 0 : ((aw < 109) ? 1 : 2));
  }
  __syncthreads();

  // ---------------- Phase 1: QKV = Xs @ Wqkv + b (two N-halves) -------------
  constexpr float qscale = 0.17677669529663687f;  // 32^-0.5
  for (int half = 0; half < 2; ++half) {
    for (int pos = wave; pos < 63; pos += 8) {      // 7 M-tiles x 9 N-tiles
      int mi = pos / 9, nl = pos % 9, ni = half * 9 + nl;
      int col = ni * 16 + l;
      float bias = qkv_b[col];
      v8f c = {bias, bias, bias, bias, bias, bias, bias, bias};
      const bf16_t* ar = Xs + (mi * 16 + l) * XS_;
      const bf16_t* br = Wq + (nl * 16 + l) * WQS_;
#pragma unroll
      for (int kk = 0; kk < 3; ++kk)
        c = wmma_bf(frag_a(ar, kk * 32 + 8 * hi2), frag_b(br, kk * 32 + 8 * hi2), c);
      if (ni < 6) {
#pragma unroll
        for (int g = 0; g < 8; ++g) Qs[(mi * 16 + g + 8 * hi2) * QS_ + col] = f2bf(c[g] * qscale);
      } else if (ni < 12) {
#pragma unroll
        for (int g = 0; g < 8; ++g) Ks[(mi * 16 + g + 8 * hi2) * KS_ + (col - 96)] = f2bf(c[g]);
      } else {
        // V stored transposed: a lane's 8 results are 8 consecutive tokens ->
        // pack into a single aligned 16B LDS store.
        v8bf p;
#pragma unroll
        for (int g = 0; g < 8; ++g) p[g] = f2bf(c[g]);
        *(v8bf*)(Vt + (col - 192) * VTS_ + mi * 16 + 8 * hi2) = p;
      }
    }
    __syncthreads();
    if (half == 0) {  // bring in cols 144..287
      for (int i = tid; i < 96 * 144; i += 256) {
        int k = i / 144, n = i % 144;
        Wq[n * WQS_ + k] = f2bf(qkv_w[k * 288 + 144 + n]);
      }
      __syncthreads();
    }
  }

  // projection accumulators (7x6 tiles distributed over 8 waves), init = bias
  v8f acc[6];
#pragma unroll
  for (int j = 0; j < 6; ++j) {
    int pos = wave + 8 * j;
    float b = (pos < 42) ? proj_b[(pos % 6) * 16 + l] : 0.f;
    v8f t = {b, b, b, b, b, b, b, b};
    acc[j] = t;
  }

  // ---------------- Phase 2: per-head attention -----------------------------
  for (int h = 0; h < 3; ++h) {
    // logits S = Q @ K^T (K-dim = 32 -> one WMMA per 16x16 tile)
    for (int pos = wave; pos < 49; pos += 8) {
      int mi = pos / 7, ni = pos % 7;
      v8f c = {};
      c = wmma_bf(frag_a(Qs + (mi * 16 + l) * QS_, h * 32 + 8 * hi2),
                  frag_b(Ks + (ni * 16 + l) * KS_, h * 32 + 8 * hi2), c);
#pragma unroll
      for (int g = 0; g < 8; ++g) {
        int r = mi * 16 + g + 8 * hi2, cc = ni * 16 + l;
        if (r < 98 && cc < SST_) Sb[r * SST_ + cc] = c[g];
      }
    }
    __syncthreads();
    // softmax row-per-thread, + rel-pos bias + shift mask
    if (tid < 112) {
      bf16_t* Pr = Pb + tid * PS_;
      if (tid < 98) {
        int it = tid / 49, r2 = tid % 49, ih = r2 / 7, iw = r2 % 7;
        int rr = regS[tid];
        float* Sr = Sb + tid * SST_;
        float mx = -1e30f;
        for (int c2 = 0; c2 < 98; ++c2) {
          int jt = c2 / 49, rc = c2 % 49, jh = rc / 7, jw = rc % 7;
          int idx = ((it - jt + 1) * 13 + (ih - jh + 6)) * 13 + (iw - jw + 6);
          float v = Sr[c2] + rpb[idx * 3 + h];
          if (regS[c2] != rr) v -= 100.f;
          Sr[c2] = v;
          mx = fmaxf(mx, v);
        }
        float sum = 0.f;
        for (int c2 = 0; c2 < 98; ++c2) { float e = expf(Sr[c2] - mx); Sr[c2] = e; sum += e; }
        float inv = 1.f / sum;
        // write probabilities (+ zero pad to col 128) as packed 16B chunks
        for (int c2 = 0; c2 < 128; c2 += 8) {
          v8bf p;
#pragma unroll
          for (int u = 0; u < 8; ++u) {
            int cc = c2 + u;
            p[u] = f2bf((cc < 98) ? Sr[cc] * inv : 0.f);
          }
          *(v8bf*)(Pr + c2) = p;
        }
      } else {
        v8bf z = {};
        for (int c2 = 0; c2 < 128; c2 += 8) *(v8bf*)(Pr + c2) = z;
      }
    }
    __syncthreads();
    // OH = P @ V  (OH safely aliases consumed logits buffer)
    for (int pos = wave; pos < 14; pos += 8) {
      int mi = pos / 2, nt = pos % 2;
      v8f c = {};
      const bf16_t* ar = Pb + (mi * 16 + l) * PS_;
      const bf16_t* br = Vt + (h * 32 + nt * 16 + l) * VTS_;
#pragma unroll
      for (int kk = 0; kk < 4; ++kk)
        c = wmma_bf(frag_a(ar, kk * 32 + 8 * hi2), frag_b(br, kk * 32 + 8 * hi2), c);
#pragma unroll
      for (int g = 0; g < 8; ++g) OH[(mi * 16 + g + 8 * hi2) * OHS_ + nt * 16 + l] = f2bf(c[g]);
    }
    __syncthreads();
    // partial projection: acc += OH @ proj_w[h*32:(h+1)*32, :]  (K=32)
#pragma unroll
    for (int j = 0; j < 6; ++j) {
      int pos = wave + 8 * j;
      if (pos < 42) {  // wave-uniform branch: EXEC stays all-ones for WMMA
        int mi = pos / 6, ni = pos % 6;
        acc[j] = wmma_bf(frag_a(OH + (mi * 16 + l) * OHS_, 8 * hi2),
                         frag_b(Wp + (ni * 16 + l) * WPS_, h * 32 + 8 * hi2), acc[j]);
      }
    }
    __syncthreads();
  }

  // ---------------- Phase 3: scatter x1 = x + proj(attn) --------------------
#pragma unroll
  for (int j = 0; j < 6; ++j) {
    int pos = wave + 8 * j;
    if (pos < 42) {
      int mi = pos / 6, col = (pos % 6) * 16 + l;
#pragma unroll
      for (int g = 0; g < 8; ++g) {
        int tok = mi * 16 + g + 8 * hi2;
        if (tok < 98) {
          int it = tok / 49, r2 = tok % 49, ih = r2 / 7, iw = r2 % 7;
          int ts = (tb * 2 + it + 1) & 15;
          int hs = hb * 7 + ih + 3; if (hs >= 112) hs -= 112;
          int vs = wb * 7 + iw + 3; if (vs >= 112) vs -= 112;
          int gi = ((ts * 112 + hs) * 112 + vs) * 96 + col;
          out[gi] = x[gi] + acc[j][g];
        }
      }
    }
  }
}

// ---------------------------------------------------------------------------
// Kernel 2: MLP (in-place on x1): out = x1 + gelu(ln2(x1)@fc1+b1)@fc2+b2
// 64 tokens per block, 3136 blocks, 256 threads.
// ---------------------------------------------------------------------------
constexpr int MXS = 104, MHS = 392, F1S = 104, F2S = 392;
constexpr int MOFF_X = 0;                       // 64*104*2 = 13312
constexpr int MOFF_H = 13312;                   // 64*392*2 = 50176
constexpr int MOFF_W = 63488;                   // max(384*104*2, 96*392*2) = 79872
constexpr int MLP_LDS = MOFF_W + 79872;         // 143360
static_assert(MLP_LDS <= 163840, "mlp LDS");

__global__ __launch_bounds__(256, 1)
void swin_mlp(const float* __restrict__ g2, const float* __restrict__ b2,
              const float* __restrict__ fc1_w, const float* __restrict__ fc1_b,
              const float* __restrict__ fc2_w, const float* __restrict__ fc2_b,
              float* __restrict__ io)
{
  extern __shared__ __align__(16) char smem[];
  bf16_t* Xs = (bf16_t*)(smem + MOFF_X);
  bf16_t* Hs = (bf16_t*)(smem + MOFF_H);
  bf16_t* Wt = (bf16_t*)(smem + MOFF_W);
  const int tid = threadIdx.x, wave = tid >> 5, lane = tid & 31, l = lane & 15, hi2 = lane >> 4;
  const int row0 = blockIdx.x * 64;

  if (tid == 0) {
    __builtin_prefetch(fc1_w, 0, 3);
    __builtin_prefetch(fc2_w, 0, 3);
  }

  // LN2 (reads x1 written by attention kernel)
  for (int t = wave; t < 64; t += 8) {
    const float* xr = io + (row0 + t) * 96;
    float v0 = xr[lane], v1 = xr[lane + 32], v2 = xr[lane + 64];
    float s = v0 + v1 + v2, q = v0 * v0 + v1 * v1 + v2 * v2;
    for (int o = 16; o >= 1; o >>= 1) { s += __shfl_xor(s, o); q += __shfl_xor(q, o); }
    float mean = s * (1.f / 96.f);
    float rstd = rsqrtf(q * (1.f / 96.f) - mean * mean + 1e-5f);
    Xs[t * MXS + lane]      = f2bf((v0 - mean) * rstd * g2[lane]      + b2[lane]);
    Xs[t * MXS + lane + 32] = f2bf((v1 - mean) * rstd * g2[lane + 32] + b2[lane + 32]);
    Xs[t * MXS + lane + 64] = f2bf((v2 - mean) * rstd * g2[lane + 64] + b2[lane + 64]);
  }
  // fc1 weights transposed: Wt[n][k] = fc1_w[k][n]
  for (int i = tid; i < 96 * 384; i += 256) {
    int k = i / 384, n = i % 384;
    Wt[n * F1S + k] = f2bf(fc1_w[i]);
  }
  __syncthreads();

  // H = gelu(X @ fc1 + b)   (4 x 24 tiles, K=96)
  for (int pos = wave; pos < 96; pos += 8) {
    int mi = pos / 24, ni = pos % 24;
    int col = ni * 16 + l;
    float bias = fc1_b[col];
    v8f c = {bias, bias, bias, bias, bias, bias, bias, bias};
    const bf16_t* ar = Xs + (mi * 16 + l) * MXS;
    const bf16_t* br = Wt + col * F1S;
#pragma unroll
    for (int kk = 0; kk < 3; ++kk)
      c = wmma_bf(frag_a(ar, kk * 32 + 8 * hi2), frag_b(br, kk * 32 + 8 * hi2), c);
#pragma unroll
    for (int g = 0; g < 8; ++g) {
      float v = c[g];
      v = 0.5f * v * (1.f + erff(v * 0.70710678118654752f));  // exact GELU
      Hs[(mi * 16 + g + 8 * hi2) * MHS + col] = f2bf(v);
    }
  }
  __syncthreads();
  // fc2 weights transposed: Wt[n][k] = fc2_w[k][n]
  for (int i = tid; i < 384 * 96; i += 256) {
    int k = i / 96, n = i % 96;
    Wt[n * F2S + k] = f2bf(fc2_w[i]);
  }
  __syncthreads();

  // out = x1 + H @ fc2 + b   (4 x 6 tiles, K=384)
  for (int pos = wave; pos < 24; pos += 8) {
    int mi = pos / 6, ni = pos % 6;
    int col = ni * 16 + l;
    float bias = fc2_b[col];
    v8f c = {bias, bias, bias, bias, bias, bias, bias, bias};
    const bf16_t* ar = Hs + (mi * 16 + l) * MHS;
    const bf16_t* br = Wt + col * F2S;
#pragma unroll
    for (int kk = 0; kk < 12; ++kk)
      c = wmma_bf(frag_a(ar, kk * 32 + 8 * hi2), frag_b(br, kk * 32 + 8 * hi2), c);
#pragma unroll
    for (int g = 0; g < 8; ++g) {
      int gi = (row0 + mi * 16 + g + 8 * hi2) * 96 + col;
      io[gi] = io[gi] + c[g];
    }
  }
}

// ---------------------------------------------------------------------------
extern "C" void kernel_launch(void* const* d_in, const int* in_sizes, int n_in,
                              void* d_out, int out_size, void* d_ws, size_t ws_size,
                              hipStream_t stream) {
  (void)in_sizes; (void)n_in; (void)out_size; (void)d_ws; (void)ws_size;
  const float* x      = (const float*)d_in[0];
  const float* g1     = (const float*)d_in[1];
  const float* b1     = (const float*)d_in[2];
  const float* qkv_w  = (const float*)d_in[3];
  const float* qkv_b  = (const float*)d_in[4];
  const float* rpb    = (const float*)d_in[5];
  const float* proj_w = (const float*)d_in[6];
  const float* proj_b = (const float*)d_in[7];
  const float* g2     = (const float*)d_in[8];
  const float* b2     = (const float*)d_in[9];
  const float* fc1_w  = (const float*)d_in[10];
  const float* fc1_b  = (const float*)d_in[11];
  const float* fc2_w  = (const float*)d_in[12];
  const float* fc2_b  = (const float*)d_in[13];
  float* out = (float*)d_out;

  // opt in to >64KB dynamic LDS (host-side attribute, graph-capture safe)
  (void)hipFuncSetAttribute((const void*)swin_attn,
                            hipFuncAttributeMaxDynamicSharedMemorySize, ATTN_LDS);
  (void)hipFuncSetAttribute((const void*)swin_mlp,
                            hipFuncAttributeMaxDynamicSharedMemorySize, MLP_LDS);

  // 2048 windows (8 x 16 x 16), one workgroup each
  swin_attn<<<dim3(2048), dim3(256), ATTN_LDS, stream>>>(
      x, g1, b1, qkv_w, qkv_b, rpb, proj_w, proj_b, out);
  // 200704 tokens / 64 per block
  swin_mlp<<<dim3(3136), dim3(256), MLP_LDS, stream>>>(
      g2, b2, fc1_w, fc1_b, fc2_w, fc2_b, out);
}
